// CausalAttentionKV_30880814858943
// MI455X (gfx1250) — compile-verified, matrix operand order checked by reading, not run on hardware
//
#include <hip/hip_runtime.h>
#include <hip/hip_bf16.h>

// ---------------------------------------------------------------------------
// CausalAttentionKV for MI455X (gfx1250), bf16 WMMA pipeline + TDM staging.
//
//   1) cvt_f32_bf16:      x (f32) -> xb (bf16)
//   2) transpose_cvt x4:  W{q,k,v,o} (f32, KxN) -> bf16 NxK (pre-transposed)
//   3) gemm_tdm:          k_all = xb @ Wk      ([b][t][d])
//   4) gemm_tdm:          vT    = xb @ Wv      (TRANSPOSED [b][d][t])
//   5) gemm_gather:       q_sel = xb[qidx]@Wq  ([b][m][d])
//   6) attn_kernel:       flash-style online softmax, v_wmma everywhere,
//                         T-loop bounded by sorted qidx per 16-query tile
//   7) gemm_tdm<f32 out>: out = y @ Wo + bo
//
// gemm_tdm: A tile (64 rows x 256 K, bf16) staged into LDS by the Tensor
// Data Mover (wave 0 issues tensor_load_to_lds; TENSORcnt wait + barrier),
// double buffered. TDM pad: +16B per 512B row => LDS row stride 528B, so the
// 16 lanes of an A-fragment read 16 distinct bank groups (528/4 mod 64 = 4).
// Each wave computes 32x64 of output: 8 v_wmma per K=32 step vs 12 b128 loads.
//
// WMMA fragment conventions (CDNA5 ISA 7.12.2, wave32):
//   A (16xK rows):  lane l (r=l&15, half=l>>4) holds row r,
//                   K elems {half*8..+7} and {16+half*8..+7}  -> 2x b128
//   B (Kx16 cols):  lane holds column r, same K pattern       -> 2x b128
//   C/D (16x16):    VGPR j, lane l -> M = j + 8*half, N = l&15
// ---------------------------------------------------------------------------

#define D_MODEL 2048
#define N_HEAD  16
#define HD      128
#define BATCH   4
#define SEQ     4096
#define MQ      512

#define KCHUNK        256               // K elems staged per TDM transfer
#define LDS_ROW_BYTES 528               // 512 data + 16 TDM pad

typedef __bf16 bf16_t;
typedef __attribute__((ext_vector_type(8)))  float        v8f;
typedef __attribute__((ext_vector_type(8)))  __bf16       v8bf;
typedef __attribute__((ext_vector_type(16))) __bf16       v16bf;
typedef __attribute__((ext_vector_type(4)))  unsigned int u32x4;
typedef __attribute__((ext_vector_type(4)))  int          i32x4;
typedef __attribute__((ext_vector_type(8)))  int          i32x8;

union BFrag { v16bf v; v8bf h[2]; };

__device__ __forceinline__ v8f wmma_bf16(v16bf a, v16bf b, v8f c) {
  return __builtin_amdgcn_wmma_f32_16x16x32_bf16(false, a, false, b,
                                                 (short)0, c, false, false);
}

// ---------------------------------------------------------------------------
// TDM: 2D bf16 tile load, global -> LDS, with LDS row padding.
//   tile: tile_d1 rows x tile_d0 elems; row stride in memory = stride0 elems.
//   pad codes: interval 6 => every 128 DWORDs (512B), amount 3 => 4 DWORDs.
//   This toolchain exposes the 6-arg builtin:
//     (u32x4 g0, i32x8 g1, i32x4 g2, i32x4 g3, i32x8 g4, i32 cpol)
// ---------------------------------------------------------------------------
#if __has_builtin(__builtin_amdgcn_tensor_load_to_lds)
#define HAVE_TDM 1
__device__ __forceinline__ void tdm_load_2d_bf16(unsigned lds_addr,
                                                 const void* gaddr,
                                                 unsigned tensor_d0,
                                                 unsigned tensor_d1,
                                                 unsigned tile_d0,
                                                 unsigned tile_d1,
                                                 unsigned stride0,
                                                 unsigned pad_iv_code,
                                                 unsigned pad_amt_code) {
  unsigned long long ga = (unsigned long long)gaddr;
  u32x4 g0;
  g0[0] = 1u;                                       // count=1, user descriptor
  g0[1] = lds_addr;                                 // lds_addr  [63:32]
  g0[2] = (unsigned)(ga & 0xFFFFFFFFu);             // global_addr lo
  g0[3] = (unsigned)((ga >> 32) & 0x1FFFFFFu)       // global_addr hi (57-bit)
        | (2u << 30);                               // type = 2 (image)
  unsigned flags = (1u << 16);                      // data_size = 1 (2 bytes)
  if (pad_amt_code || pad_iv_code)
    flags |= (1u << 20) | (pad_iv_code << 22) | (pad_amt_code << 25);
  i32x8 g1;
  g1[0] = (int)flags;
  g1[1] = (int)((tensor_d0 & 0xFFFFu) << 16);       // barrier_addr=0 | d0.lo16
  g1[2] = (int)((tensor_d0 >> 16) | ((tensor_d1 & 0xFFFFu) << 16));
  g1[3] = (int)(((tensor_d1 >> 16) & 0xFFFFu) | (tile_d0 << 16));
  g1[4] = (int)(tile_d1 & 0xFFFFu);                 // tile_d1 | tile_d2=0
  g1[5] = (int)stride0;                             // dim0 stride lo32
  g1[6] = 0;                                        // dim0 stride hi16
  g1[7] = 0;
  i32x4 z4 = {0, 0, 0, 0};
  i32x8 z8 = {0, 0, 0, 0, 0, 0, 0, 0};
  __builtin_amdgcn_tensor_load_to_lds(g0, g1, z4, z4, z8, 0);
}
#else
#define HAVE_TDM 0
#endif

// Stage one 64 x KCHUNK A tile into (padded) LDS.
__device__ __forceinline__ void stage_A(char* ldsDst, const bf16_t* gsrc,
                                        int lda, int tid, int wave) {
#if HAVE_TDM
  if (wave == 0) {
    tdm_load_2d_bf16((unsigned)(size_t)ldsDst, (const void*)gsrc,
                     (unsigned)lda, 1u << 20, KCHUNK, 64, (unsigned)lda,
                     /*pad every 512B*/ 6, /*by 16B*/ 3);
  }
#else
  // cooperative fallback: 64 rows x 512B = 2048 x 16B chunks, 256 threads
#pragma unroll
  for (int i = 0; i < 8; ++i) {
    const int c   = tid + i * 256;
    const int row = c >> 5;
    const int c16 = c & 31;
    *(v8bf*)(ldsDst + row * LDS_ROW_BYTES + c16 * 16) =
        *(const v8bf*)(gsrc + (long)row * lda + c16 * 8);
  }
#endif
  (void)tid; (void)wave; (void)lda;
}

__device__ __forceinline__ void stage_A_wait(int wave) {
#if HAVE_TDM
  if (wave == 0) __builtin_amdgcn_s_wait_tensorcnt(0);
#endif
  (void)wave;
}

// ---------------------------------------------------------------------------
// 1) f32 -> bf16 bulk convert
// ---------------------------------------------------------------------------
__global__ void cvt_f32_bf16_kernel(const float* __restrict__ src,
                                    bf16_t* __restrict__ dst, long n) {
  long i = ((long)blockIdx.x * blockDim.x + threadIdx.x) * 8;
  if (i + 8 > n) return;
  float4 f0 = *(const float4*)(src + i);
  float4 f1 = *(const float4*)(src + i + 4);
  v8bf o;
  o[0] = (bf16_t)f0.x; o[1] = (bf16_t)f0.y; o[2] = (bf16_t)f0.z; o[3] = (bf16_t)f0.w;
  o[4] = (bf16_t)f1.x; o[5] = (bf16_t)f1.y; o[6] = (bf16_t)f1.z; o[7] = (bf16_t)f1.w;
  *(v8bf*)(dst + i) = o;
}

// ---------------------------------------------------------------------------
// 2) W (f32, [k][n]) -> WT (bf16, [n][k]), LDS-tiled transpose
// ---------------------------------------------------------------------------
__global__ void transpose_cvt_kernel(const float* __restrict__ W,
                                     bf16_t* __restrict__ WT) {
  __shared__ float tile[32][33];
  const int bc = blockIdx.x * 32;
  const int br = blockIdx.y * 32;
  const int tx = threadIdx.x, ty = threadIdx.y;
#pragma unroll
  for (int i = 0; i < 32; i += 8)
    tile[ty + i][tx] = W[(long)(br + ty + i) * D_MODEL + bc + tx];
  __syncthreads();
#pragma unroll
  for (int i = 0; i < 32; i += 8)
    WT[(long)(bc + ty + i) * D_MODEL + br + tx] = (bf16_t)tile[tx][ty + i];
}

// ---------------------------------------------------------------------------
// 3,4,7) TDM-staged WMMA GEMM: C[row,col] = A[row,:] . BT[col,:] + bias
//   Block = 256 thr = 8 waves as 2(M) x 4(N); block tile 64 x 256.
//   Wave tile 32 x 64: acc[2][4], B-frag reused by 2 wmma.
//   A staged in LDS by TDM, double-buffered per 256-K chunk.
// ---------------------------------------------------------------------------
template <bool OUTF32>
__global__ __launch_bounds__(256) void gemm_tdm_bf16_kernel(
    const bf16_t* __restrict__ A, long aBatch, int lda,
    const bf16_t* __restrict__ BT, int ldb,
    const float* __restrict__ bias,
    void* __restrict__ outp, long oBatch, long oRow, long oCol, int K) {
  __shared__ __align__(128) char ldsA[2][64 * LDS_ROW_BYTES];

  const int tid  = threadIdx.x;
  const int w    = tid >> 5;
  const int lane = tid & 31;
  const int half = lane >> 4;
  const int r    = lane & 15;
  const int wm   = w >> 2;                   // 0..1 : M sub-block
  const int wn   = w & 3;                    // 0..3 : N sub-block
  const int row0 = blockIdx.y * 64;
  const int col0 = blockIdx.x * 256 + wn * 64;
  const int z    = blockIdx.z;

  const bf16_t* Ab = A + (long)z * aBatch + (long)row0 * lda;

  const bf16_t* bp[4];
#pragma unroll
  for (int s = 0; s < 4; ++s) bp[s] = BT + (long)(col0 + s * 16 + r) * ldb;

  v8f zf = {};
  v8f acc[2][4];
#pragma unroll
  for (int m = 0; m < 2; ++m)
#pragma unroll
    for (int s = 0; s < 4; ++s) acc[m][s] = zf;

  const int nk = K / KCHUNK;

  stage_A(ldsA[0], Ab, lda, tid, w);
  stage_A_wait(w);
  __syncthreads();

  for (int kc = 0; kc < nk; ++kc) {
    const int cur = kc & 1;
    if (kc + 1 < nk) {
      stage_A(ldsA[1 - cur], Ab + (long)(kc + 1) * KCHUNK, lda, tid, w);
#pragma unroll
      for (int s = 0; s < 4; ++s)   // pull next weight chunk toward GL2
        __builtin_prefetch((const void*)(bp[s] + (long)(kc + 1) * KCHUNK), 0, 1);
    }

    const char* abase = ldsA[cur];
#pragma unroll
    for (int k32 = 0; k32 < KCHUNK / 32; ++k32) {
      const int klo = k32 * 32 + half * 8;
      const int khi = klo + 16;
      BFrag af0, af1;
      af0.h[0] = *(const v8bf*)(abase + (wm * 32 + r)      * LDS_ROW_BYTES + klo * 2);
      af0.h[1] = *(const v8bf*)(abase + (wm * 32 + r)      * LDS_ROW_BYTES + khi * 2);
      af1.h[0] = *(const v8bf*)(abase + (wm * 32 + 16 + r) * LDS_ROW_BYTES + klo * 2);
      af1.h[1] = *(const v8bf*)(abase + (wm * 32 + 16 + r) * LDS_ROW_BYTES + khi * 2);
      const long kg = (long)kc * KCHUNK + klo;
#pragma unroll
      for (int s = 0; s < 4; ++s) {
        BFrag bf;
        bf.h[0] = *(const v8bf*)(bp[s] + kg);
        bf.h[1] = *(const v8bf*)(bp[s] + kg + 16);
        acc[0][s] = wmma_bf16(af0.v, bf.v, acc[0][s]);
        acc[1][s] = wmma_bf16(af1.v, bf.v, acc[1][s]);
      }
    }

    if (kc + 1 < nk) {
      stage_A_wait(w);     // wave 0 drains TENSORcnt before releasing barrier
      __syncthreads();
    }
  }

  // epilogue
#pragma unroll
  for (int s = 0; s < 4; ++s) {
    const int col = col0 + s * 16 + r;
    const float bs = bias ? bias[col] : 0.0f;
#pragma unroll
    for (int m = 0; m < 2; ++m) {
#pragma unroll
      for (int j = 0; j < 8; ++j) {
        const long row = row0 + wm * 32 + m * 16 + 8 * half + j;
        const long idx = (long)z * oBatch + row * oRow + (long)col * oCol;
        const float v  = acc[m][s][j] + bs;
        if (OUTF32) ((float*)outp)[idx]  = v;
        else        ((bf16_t*)outp)[idx] = (bf16_t)v;
      }
    }
  }
}

// ---------------------------------------------------------------------------
// 5) Direct-load WMMA GEMM with row gather (for q_sel; M is small/irregular).
//   Wave computes 16x32; block tile 16 x 256.
// ---------------------------------------------------------------------------
__global__ __launch_bounds__(256) void gemm_gather_bf16_kernel(
    const bf16_t* __restrict__ A, long aBatch, int lda,
    const bf16_t* __restrict__ BT, int ldb,
    const int* __restrict__ gidx, int gBatch,
    const float* __restrict__ bias,
    bf16_t* __restrict__ outp, long oBatch, int ldo, int K) {
  const int w    = threadIdx.x >> 5;
  const int lane = threadIdx.x & 31;
  const int half = lane >> 4;
  const int r    = lane & 15;
  const int row0 = blockIdx.y * 16;
  const int col0 = blockIdx.x * 256 + w * 32;
  const int z    = blockIdx.z;

  const long arow   = (long)gidx[(long)z * gBatch + row0 + r];
  const bf16_t* ap  = A + (long)z * aBatch + arow * (long)lda;
  const bf16_t* bp0 = BT + (long)(col0 + r) * ldb;
  const bf16_t* bp1 = BT + (long)(col0 + 16 + r) * ldb;

  v8f acc0 = {}, acc1 = {};
  for (int k0 = 0; k0 < K; k0 += 32) {
    const int klo = k0 + half * 8;
    const int khi = klo + 16;
    BFrag a, b0, b1;
    a.h[0]  = *(const v8bf*)(ap  + klo);  a.h[1]  = *(const v8bf*)(ap  + khi);
    b0.h[0] = *(const v8bf*)(bp0 + klo);  b0.h[1] = *(const v8bf*)(bp0 + khi);
    b1.h[0] = *(const v8bf*)(bp1 + klo);  b1.h[1] = *(const v8bf*)(bp1 + khi);
    acc0 = wmma_bf16(a.v, b0.v, acc0);
    acc1 = wmma_bf16(a.v, b1.v, acc1);
  }
  const float bias0 = bias ? bias[col0 + r]      : 0.0f;
  const float bias1 = bias ? bias[col0 + 16 + r] : 0.0f;
#pragma unroll
  for (int j = 0; j < 8; ++j) {
    const long row = row0 + 8 * half + j;
    outp[(long)z * oBatch + row * ldo + col0 + r]      = (bf16_t)(acc0[j] + bias0);
    outp[(long)z * oBatch + row * ldo + col0 + 16 + r] = (bf16_t)(acc1[j] + bias1);
  }
}

// ---------------------------------------------------------------------------
// 6) Attention: one wave per 16-query tile of one (b,h). Flash-style online
//    softmax; probs C->A layout transposed via per-wave LDS tile (same-wave
//    LDS ops are in-order, no block barrier across divergent T-loops).
// ---------------------------------------------------------------------------
__global__ __launch_bounds__(256) void attn_kernel(
    const bf16_t* __restrict__ q_sel,  // [B][MQ][D_MODEL]
    const bf16_t* __restrict__ k_all,  // [B][SEQ][D_MODEL]
    const bf16_t* __restrict__ vT,     // [B][D_MODEL][SEQ]
    const int*    __restrict__ qix,    // [B][MQ], sorted ascending
    bf16_t*       __restrict__ yout)   // [B][MQ][D_MODEL]
{
  const int w    = threadIdx.x >> 5;
  const int lane = threadIdx.x & 31;
  const int half = lane >> 4;
  const int r    = lane & 15;
  const int m0   = blockIdx.x * 128 + w * 16;
  const int h    = blockIdx.y;
  const int b    = blockIdx.z;

  __shared__ __align__(128) bf16_t probs[8][16][32];

  BFrag qf[4];
  const bf16_t* qb = q_sel + ((long)(b * MQ + m0 + r)) * D_MODEL + h * HD;
#pragma unroll
  for (int kc = 0; kc < 4; ++kc) {
    const int klo = kc * 32 + half * 8;
    qf[kc].h[0] = *(const v8bf*)(qb + klo);
    qf[kc].h[1] = *(const v8bf*)(qb + klo + 16);
  }

  int qv[8];
#pragma unroll
  for (int j = 0; j < 8; ++j) qv[j] = qix[b * MQ + m0 + 8 * half + j];
  const int t_hi = qix[b * MQ + m0 + 15];

  float m_run[8], l_run[8], alpha[8];
  v8f zf = {};
  v8f acc[8];
#pragma unroll
  for (int n = 0; n < 8; ++n) acc[n] = zf;
#pragma unroll
  for (int j = 0; j < 8; ++j) { m_run[j] = -1e30f; l_run[j] = 0.0f; }

  const float SCALE = 0.08838834764831845f;  // 1/sqrt(HD)

  for (int t0 = 0; t0 <= t_hi; t0 += 32) {
    v8f s0 = zf, s1 = zf;
    {
      const bf16_t* kb = k_all + ((long)(b * SEQ + t0 + r)) * D_MODEL + h * HD;
#pragma unroll
      for (int kc = 0; kc < 4; ++kc) {
        const int klo = kc * 32 + half * 8;
        BFrag kf;
        kf.h[0] = *(const v8bf*)(kb + klo);
        kf.h[1] = *(const v8bf*)(kb + klo + 16);
        s0 = wmma_bf16(qf[kc].v, kf.v, s0);
      }
    }
    {
      const bf16_t* kb = k_all + ((long)(b * SEQ + t0 + 16 + r)) * D_MODEL + h * HD;
#pragma unroll
      for (int kc = 0; kc < 4; ++kc) {
        const int klo = kc * 32 + half * 8;
        BFrag kf;
        kf.h[0] = *(const v8bf*)(kb + klo);
        kf.h[1] = *(const v8bf*)(kb + klo + 16);
        s1 = wmma_bf16(qf[kc].v, kf.v, s1);
      }
    }

#pragma unroll
    for (int j = 0; j < 8; ++j) {
      float x0 = (t0 + r      <= qv[j]) ? (float)s0[j] * SCALE : -1e30f;
      float x1 = (t0 + 16 + r <= qv[j]) ? (float)s1[j] * SCALE : -1e30f;
      float bm = fmaxf(x0, x1);
#pragma unroll
      for (int off = 1; off < 16; off <<= 1)
        bm = fmaxf(bm, __shfl_xor(bm, off, 32));
      const float mn = fmaxf(m_run[j], bm);
      alpha[j] = __expf(m_run[j] - mn);
      const float p0 = __expf(x0 - mn);
      const float p1 = __expf(x1 - mn);
      float ps = p0 + p1;
#pragma unroll
      for (int off = 1; off < 16; off <<= 1)
        ps += __shfl_xor(ps, off, 32);
      l_run[j] = l_run[j] * alpha[j] + ps;
      m_run[j] = mn;
      probs[w][8 * half + j][r]      = (bf16_t)p0;
      probs[w][8 * half + j][16 + r] = (bf16_t)p1;
    }

#pragma unroll
    for (int n = 0; n < 8; ++n)
#pragma unroll
      for (int j = 0; j < 8; ++j) acc[n][j] = acc[n][j] * alpha[j];

    __builtin_amdgcn_fence(__ATOMIC_SEQ_CST, "wavefront");

    BFrag pa;
    pa.h[0] = *(const v8bf*)&probs[w][r][half * 8];
    pa.h[1] = *(const v8bf*)&probs[w][r][16 + half * 8];

#pragma unroll
    for (int n = 0; n < 8; ++n) {
      const bf16_t* vb =
          vT + ((long)b * D_MODEL + h * HD + n * 16 + r) * SEQ + t0 + half * 8;
      BFrag vf;
      vf.h[0] = *(const v8bf*)vb;
      vf.h[1] = *(const v8bf*)(vb + 16);
      acc[n] = wmma_bf16(pa.v, vf.v, acc[n]);
    }
  }

#pragma unroll
  for (int n = 0; n < 8; ++n) {
#pragma unroll
    for (int j = 0; j < 8; ++j) {
      const long row = m0 + 8 * half + j;
      const long col = h * HD + n * 16 + r;
      yout[((long)(b * MQ) + row) * D_MODEL + col] =
          (bf16_t)(acc[n][j] / l_run[j]);
    }
  }
}

// ---------------------------------------------------------------------------
// Host-side orchestration
// ---------------------------------------------------------------------------
extern "C" void kernel_launch(void* const* d_in, const int* in_sizes, int n_in,
                              void* d_out, int out_size, void* d_ws,
                              size_t ws_size, hipStream_t stream) {
  (void)in_sizes; (void)n_in; (void)out_size; (void)ws_size;

  const float* x    = (const float*)d_in[0];
  const int*   qidx = (const int*)d_in[1];
  const float* Wq   = (const float*)d_in[2];
  const float* bq   = (const float*)d_in[3];
  const float* Wk   = (const float*)d_in[4];
  const float* bk   = (const float*)d_in[5];
  const float* Wv   = (const float*)d_in[6];
  const float* bv   = (const float*)d_in[7];
  const float* Wo   = (const float*)d_in[8];
  const float* bo   = (const float*)d_in[9];
  float* out = (float*)d_out;

  const long XB = (long)BATCH * SEQ * D_MODEL;
  const long WE = (long)D_MODEL * D_MODEL;
  const long QE = (long)BATCH * MQ * D_MODEL;
  char* ws = (char*)d_ws;
  bf16_t* xb   = (bf16_t*)ws;                     // [B][T][D]
  bf16_t* wqT  = (bf16_t*)(ws + 2 * XB);          // [N][K] x4
  bf16_t* wkT  = wqT + WE;
  bf16_t* wvT  = wkT + WE;
  bf16_t* woT  = wvT + WE;
  bf16_t* kall = woT + WE;                        // [B][T][D]
  bf16_t* vTr  = kall + XB;                       // [B][D][T]
  bf16_t* qsel = vTr + XB;                        // [B][M][D]
  bf16_t* yb   = qsel + QE;                       // [B][M][D]

  cvt_f32_bf16_kernel<<<(int)(XB / 8 / 256), 256, 0, stream>>>(x, xb, XB);

  dim3 tb(32, 8), tg(D_MODEL / 32, D_MODEL / 32);
  transpose_cvt_kernel<<<tg, tb, 0, stream>>>(Wq, wqT);
  transpose_cvt_kernel<<<tg, tb, 0, stream>>>(Wk, wkT);
  transpose_cvt_kernel<<<tg, tb, 0, stream>>>(Wv, wvT);
  transpose_cvt_kernel<<<tg, tb, 0, stream>>>(Wo, woT);

  const long sBatchA = (long)SEQ * D_MODEL;

  // K projection -> k_all [b][t][d]
  gemm_tdm_bf16_kernel<false>
      <<<dim3(D_MODEL / 256, SEQ / 64, BATCH), 256, 0, stream>>>(
          xb, sBatchA, D_MODEL, wkT, D_MODEL, bk,
          kall, sBatchA, D_MODEL, 1, D_MODEL);

  // V projection -> vT [b][d][t] (transposed output for P.V B-fragments)
  gemm_tdm_bf16_kernel<false>
      <<<dim3(D_MODEL / 256, SEQ / 64, BATCH), 256, 0, stream>>>(
          xb, sBatchA, D_MODEL, wvT, D_MODEL, bv,
          vTr, (long)D_MODEL * SEQ, 1, SEQ, D_MODEL);

  // Q projection on gathered rows -> q_sel [b][m][d]
  gemm_gather_bf16_kernel
      <<<dim3(D_MODEL / 256, MQ / 16, BATCH), 256, 0, stream>>>(
          xb, sBatchA, D_MODEL, wqT, D_MODEL, qidx, MQ, bq,
          qsel, (long)MQ * D_MODEL, D_MODEL, D_MODEL);

  // attention
  attn_kernel<<<dim3(MQ / 128, N_HEAD, BATCH), 256, 0, stream>>>(
      qsel, kall, vTr, qidx, yb);

  // output projection -> f32 d_out
  gemm_tdm_bf16_kernel<true>
      <<<dim3(D_MODEL / 256, (BATCH * MQ) / 64, 1), 256, 0, stream>>>(
          yb, 0, D_MODEL, woT, D_MODEL, bo,
          out, 0, D_MODEL, 1, D_MODEL);
}
